// VQVAE_68255620268373
// MI455X (gfx1250) — matrix operand order repaired and checked by reading, not compiled, not run on hardware
//
#include <hip/hip_runtime.h>
#include <cstdint>

// Problem dims
#define NROWS 32768   // B*T = 16*2048
#define D_IN  80
#define DD    512
#define KC    1024

typedef float  v2f   __attribute__((ext_vector_type(2)));
typedef float  v8f   __attribute__((ext_vector_type(8)));
typedef __bf16 v16bf __attribute__((ext_vector_type(16)));

__device__ __forceinline__ unsigned short f2bf(float f) {
    unsigned int u = __float_as_uint(f);
    u = (u + 0x7FFFu + ((u >> 16) & 1u)) >> 16;   // round-to-nearest-even
    return (unsigned short)u;
}
__device__ __forceinline__ float bf2f(unsigned short h) {
    return __uint_as_float(((unsigned int)h) << 16);
}

// ---------------------------------------------------------------------------
// K0: centroids fp32 -> bf16, and cnorm[k] = |bf16(centroid_k)|^2
// ---------------------------------------------------------------------------
__global__ void k0_centroids(const float* __restrict__ cent,
                             unsigned short* __restrict__ cbf,
                             float* __restrict__ cnorm) {
    int row = blockIdx.x;
    int tid = threadIdx.x;
    const float* src = cent + row * DD;
    unsigned short* dst = cbf + row * DD;
    int i = tid * 2;
    unsigned short h0 = f2bf(src[i]);
    unsigned short h1 = f2bf(src[i + 1]);
    ((unsigned int*)dst)[tid] = (unsigned int)h0 | ((unsigned int)h1 << 16);
    float a = bf2f(h0), b = bf2f(h1);
    float s = a * a + b * b;
    for (int m = 16; m >= 1; m >>= 1) s += __shfl_xor(s, m, 32);
    __shared__ float red[8];
    int w = tid >> 5, l = tid & 31;
    if (l == 0) red[w] = s;
    __syncthreads();
    if (tid == 0) {
        float t = 0.f;
        for (int j = 0; j < 8; ++j) t += red[j];
        cnorm[row] = t;
    }
}

// ---------------------------------------------------------------------------
// K1: encoder x[N,80] @ W_enc[80,512] + b_enc -> bf16 x_enc[N,512]
// fp32 WMMA 16x16x4; one wave does 16 rows x 64 cols (4 tiles).
// ---------------------------------------------------------------------------
__global__ __launch_bounds__(256) void k1_encoder(const float* __restrict__ x,
                           const float* __restrict__ Wenc,
                           const float* __restrict__ benc,
                           unsigned short* __restrict__ xenc) {
    int wave = blockIdx.x * (blockDim.x >> 5) + (threadIdx.x >> 5);
    int lane = threadIdx.x & 31;
    int mt = wave >> 3;           // 2048 row tiles
    int ng = wave & 7;            // 8 groups of 64 cols
    int row0 = mt * 16;
    int c0   = ng * 64;
    int g  = lane >> 4;
    int mr = lane & 15;

    const float* arow = x + (size_t)(row0 + mr) * D_IN;
    v8f acc[4];
    for (int t = 0; t < 4; ++t) acc[t] = (v8f){0.f,0.f,0.f,0.f,0.f,0.f,0.f,0.f};

    for (int k0 = 0; k0 < D_IN; k0 += 4) {
        v2f a = *(const v2f*)(arow + k0 + 2 * g);
        const float* wrow0 = Wenc + (size_t)(k0 + 2 * g) * DD;
        #pragma unroll
        for (int t = 0; t < 4; ++t) {
            int n = c0 + t * 16 + mr;
            v2f b;
            b.x = wrow0[n];
            b.y = wrow0[DD + n];
            acc[t] = __builtin_amdgcn_wmma_f32_16x16x4_f32(false, a, false, b,
                                                           (short)0, acc[t], false, false);
        }
    }
    #pragma unroll
    for (int t = 0; t < 4; ++t) {
        int n = c0 + t * 16 + mr;
        float bias = benc[n];
        #pragma unroll
        for (int r = 0; r < 8; ++r) {
            int m = row0 + r + 8 * g;               // C/D: M = r + 8*(lane/16)
            xenc[(size_t)m * DD + n] = f2bf(acc[t][r] + bias);
        }
    }
}

// ---------------------------------------------------------------------------
// K1b: xnorm[n] = |bf16 row|^2, one wave per row
// ---------------------------------------------------------------------------
__global__ void k1b_xnorm(const unsigned short* __restrict__ xenc,
                          float* __restrict__ xnorm) {
    int row  = blockIdx.x * (blockDim.x >> 5) + (threadIdx.x >> 5);
    int lane = threadIdx.x & 31;
    const uint4* q = (const uint4*)(xenc + (size_t)row * DD + lane * 16);
    uint4 q0 = q[0], q1 = q[1];
    unsigned int u[8] = {q0.x, q0.y, q0.z, q0.w, q1.x, q1.y, q1.z, q1.w};
    float s = 0.f;
    for (int j = 0; j < 8; ++j) {
        float a = bf2f((unsigned short)(u[j] & 0xFFFFu));
        float b = bf2f((unsigned short)(u[j] >> 16));
        s += a * a + b * b;
    }
    for (int m = 16; m >= 1; m >>= 1) s += __shfl_xor(s, m, 32);
    if (lane == 0) xnorm[row] = s;
}

// ---------------------------------------------------------------------------
// K2: fused distance + argmin.
//  - block = 8 waves = 128 rows; each wave keeps its entire 16x512 bf16 A
//    in registers (16 v16bf fragments, loaded once).
//  - centroid tiles staged with GLOBAL_LOAD_ASYNC_TO_LDS_B128 into a
//    double-buffered LDS tile (rows padded to 1040 B for bank spread);
//    the async DMA of tile ct+1 overlaps the 16-WMMA burst on tile ct.
// ---------------------------------------------------------------------------
#define CROW_PAD (DD + 8)     // halves; 1040-byte LDS row stride (16B aligned)

__global__ __launch_bounds__(256) void k2_argmin(const unsigned short* __restrict__ xenc,
                          const unsigned short* __restrict__ cbf,
                          const float* __restrict__ xnorm,
                          const float* __restrict__ cnorm,
                          int* __restrict__ ali) {
    __shared__ unsigned short btile[2][16 * CROW_PAD];   // 2 x 16.25 KiB

    int tid  = threadIdx.x;
    int wv   = tid >> 5;
    int lane = tid & 31;
    int g  = lane >> 4;
    int mr = lane & 15;
    int row0 = (blockIdx.x * 8 + wv) * 16;

    union F { v16bf v; uint4 q[2]; };

    // A fragments: resident for the whole kernel (128 VGPRs)
    F afr[16];
    const unsigned short* arow = xenc + (size_t)(row0 + mr) * DD;
    #pragma unroll
    for (int kk = 0; kk < 16; ++kk) {
        int k0 = kk * 32;
        afr[kk].q[0] = *(const uint4*)(arow + k0 + 8 * g);
        afr[kk].q[1] = *(const uint4*)(arow + k0 + 16 + 8 * g);
    }

    float xn[8];
    #pragma unroll
    for (int r = 0; r < 8; ++r) xn[r] = xnorm[row0 + r + 8 * g];

    float best[8];
    int   bidx[8];
    #pragma unroll
    for (int r = 0; r < 8; ++r) { best[r] = 3.4e38f; bidx[r] = 0x7FFFFFFF; }

    // per-thread LDS destinations (4 x b128 per tile) and global offsets
    int idx0 = tid * 4;                         // first of 4 uint4 chunks
    // chunk idx -> (row = idx>>6, col = idx&63); LDS byte offset row*1040's
    // worth: row * CROW_PAD * 2 + col * 16
    unsigned int lbase0 = (unsigned int)(uintptr_t)(&btile[0][0]);
    unsigned int lbase1 = (unsigned int)(uintptr_t)(&btile[1][0]);

    // issue async stage of tile `ct` into buffer `buf`
    auto stage = [&](int ct, int buf) {
        const unsigned short* gsrc = cbf + (size_t)ct * 16 * DD;
        unsigned int lb = buf ? lbase1 : lbase0;
        #pragma unroll
        for (int j = 0; j < 4; ++j) {
            int idx = idx0 + j;                 // 0..1023
            unsigned int ldsa = lb + (unsigned int)((idx >> 6) * (CROW_PAD * 2)
                                                    + (idx & 63) * 16);
            unsigned int goff = (unsigned int)(idx * 16);
            asm volatile("global_load_async_to_lds_b128 %0, %1, %2"
                         :: "v"(ldsa), "v"(goff), "s"(gsrc)
                         : "memory");
        }
    };

    stage(0, 0);
    asm volatile("s_wait_asynccnt 0x0" ::: "memory");
    __syncthreads();

    for (int ct = 0; ct < KC / 16; ++ct) {
        int cur = ct & 1;
        if (ct + 1 < KC / 16) stage(ct + 1, cur ^ 1);   // overlap DMA with WMMAs

        v8f acc = {0.f, 0.f, 0.f, 0.f, 0.f, 0.f, 0.f, 0.f};
        const unsigned short* brow = &btile[cur][mr * CROW_PAD];
        #pragma unroll
        for (int kk = 0; kk < 16; ++kk) {
            int k0 = kk * 32;
            F fb;
            fb.q[0] = *(const uint4*)(brow + k0 + 8 * g);
            fb.q[1] = *(const uint4*)(brow + k0 + 16 + 8 * g);
            acc = __builtin_amdgcn_wmma_f32_16x16x32_bf16(false, afr[kk].v, false, fb.v,
                                                          (short)0, acc, false, false);
        }

        int   nidx = ct * 16 + mr;
        float cn   = cnorm[nidx];
        #pragma unroll
        for (int r = 0; r < 8; ++r) {
            float d = xn[r] + cn - 2.0f * acc[r];
            if (d < best[r] || (d == best[r] && nidx < bidx[r])) {
                best[r] = d; bidx[r] = nidx;
            }
        }

        // next tile's async loads must be complete (and this tile's reads done
        // in all waves) before the following iteration computes / re-stages
        asm volatile("s_wait_asynccnt 0x0" ::: "memory");
        __syncthreads();
    }

    // min+argmin across the 16 lanes of each half-wave group (column dim)
    #pragma unroll
    for (int r = 0; r < 8; ++r) {
        for (int m = 8; m >= 1; m >>= 1) {
            float od = __shfl_xor(best[r], m, 32);
            int   oi = __shfl_xor(bidx[r], m, 32);
            if (od < best[r] || (od == best[r] && oi < bidx[r])) {
                best[r] = od; bidx[r] = oi;
            }
        }
        if (mr == 0) ali[row0 + r + 8 * g] = bidx[r];
    }
}

// ---------------------------------------------------------------------------
// K3: decoder. One wave = 16 rows x all 80 output cols (5 tiles).
// ---------------------------------------------------------------------------
__global__ __launch_bounds__(256) void k3_decoder(const float* __restrict__ cent,
                           const float* __restrict__ Wdec,
                           const float* __restrict__ bdec,
                           const int* __restrict__ ali,
                           float* __restrict__ out) {
    int wave = blockIdx.x * (blockDim.x >> 5) + (threadIdx.x >> 5);
    int lane = threadIdx.x & 31;
    int row0 = wave * 16;
    int g  = lane >> 4;
    int mr = lane & 15;

    const float* crow = cent + (size_t)ali[row0 + mr] * DD;
    v8f acc[5];
    for (int t = 0; t < 5; ++t) acc[t] = (v8f){0.f,0.f,0.f,0.f,0.f,0.f,0.f,0.f};

    for (int k0 = 0; k0 < DD; k0 += 4) {
        v2f a = *(const v2f*)(crow + k0 + 2 * g);
        const float* wrow0 = Wdec + (size_t)(k0 + 2 * g) * D_IN;
        #pragma unroll
        for (int t = 0; t < 5; ++t) {
            int n = t * 16 + mr;
            v2f b;
            b.x = wrow0[n];
            b.y = wrow0[D_IN + n];
            acc[t] = __builtin_amdgcn_wmma_f32_16x16x4_f32(false, a, false, b,
                                                           (short)0, acc[t], false, false);
        }
    }
    #pragma unroll
    for (int t = 0; t < 5; ++t) {
        int n = t * 16 + mr;
        float bias = bdec[n];
        #pragma unroll
        for (int r = 0; r < 8; ++r)
            out[(size_t)(row0 + r + 8 * g) * D_IN + n] = acc[t][r] + bias;
    }
}

// ---------------------------------------------------------------------------
extern "C" void kernel_launch(void* const* d_in, const int* in_sizes, int n_in,
                              void* d_out, int out_size, void* d_ws, size_t ws_size,
                              hipStream_t stream) {
    const float* x    = (const float*)d_in[0];
    const float* Wenc = (const float*)d_in[1];
    const float* benc = (const float*)d_in[2];
    const float* cent = (const float*)d_in[3];
    const float* Wdec = (const float*)d_in[4];
    const float* bdec = (const float*)d_in[5];
    float* out = (float*)d_out;

    char* ws = (char*)d_ws;
    unsigned short* xenc = (unsigned short*)(ws);                 // 32 MiB
    unsigned short* cbf  = (unsigned short*)(ws + (33554432));    // 1 MiB
    float* xnorm = (float*)(ws + 33554432 + 1048576);             // 128 KiB
    float* cnorm = (float*)(ws + 33554432 + 1048576 + 131072);    // 4 KiB
    int*   ali   = (int*)  (ws + 33554432 + 1048576 + 131072 + 4096); // 128 KiB

    k0_centroids<<<KC, 256, 0, stream>>>(cent, cbf, cnorm);
    k1_encoder  <<<(NROWS / 16), 256, 0, stream>>>(x, Wenc, benc, xenc);       // 2048*8 waves
    k1b_xnorm   <<<NROWS / 8, 256, 0, stream>>>(xenc, xnorm);
    k2_argmin   <<<NROWS / 128, 256, 0, stream>>>(xenc, cbf, xnorm, cnorm, ali);
    k3_decoder  <<<(NROWS / 16) / 8, 256, 0, stream>>>(cent, Wdec, bdec, ali, out);
}
